// MambaBlock_23527830847899
// MI455X (gfx1250) — matrix operand order repaired and checked by reading, not compiled, not run on hardware
//
#include <hip/hip_runtime.h>
#include <hip/hip_bf16.h>

// ---------------------------------------------------------------------------
// Mamba block for MI455X (gfx1250, wave32, WMMA + async global->LDS staging).
//
// Pipeline (intermediates in d_ws, ~75 MB, L2-resident):
//   0) convert x -> bf16; transpose+convert weights -> bf16 (N x K layout)
//   1) WMMA GEMM (LDS-staged B): x(2048x768) @ W_in -> x_and_res f32
//   2) fused depthwise causal conv (K=4) + SiLU -> xc (f32 + bf16)
//   3) WMMA GEMM (direct): xc @ W_x -> x_dbl (2048x80)
//   4) extract delta cols [0,48), zero-pad K to 64, bf16
//   5) WMMA GEMM (LDS-staged B) + softplus epilogue -> delta f32
//   6) fused selective-scan recurrence + D skip + silu(res) gate -> ys bf16
//   7) WMMA GEMM (LDS-staged B): ys @ W_out -> d_out f32
// ---------------------------------------------------------------------------

#define D_MODEL 768
#define D_IN    1536
#define NSTATE  16
#define DT_RANK 48
#define KCONV   4
#define B_SZ    2
#define LSEQ    1024
#define MROWS   (B_SZ * LSEQ)            // 2048
#define XDBL_W  (DT_RANK + 2 * NSTATE)   // 80
#define KPAD_DT 64

typedef __attribute__((ext_vector_type(16))) __bf16 v16bf;
typedef __attribute__((ext_vector_type(8)))  float  v8f;

struct Frag { union { v16bf v; unsigned int u[8]; }; };

// ---------------- CDNA5 async helpers ----------------

__device__ __forceinline__ void wait_asynccnt0() {
#if defined(__has_builtin) && __has_builtin(__builtin_amdgcn_s_wait_asynccnt)
    __builtin_amdgcn_s_wait_asynccnt(0);
#else
    asm volatile("s_wait_asynccnt 0" ::: "memory");
#endif
}

// One per-lane 16B async transfer global -> LDS (tracked by ASYNCcnt).
__device__ __forceinline__ void async_g2l_b128(unsigned lds_off, const void* gaddr) {
    asm volatile("global_load_async_to_lds_b128 %0, %1, off"
                 :: "v"(lds_off), "v"(gaddr)
                 : "memory");
}

// Low 32 bits of a generic LDS pointer == wave-relative LDS byte offset
// (flat aperture rule).
__device__ __forceinline__ unsigned lds_off_of(const void* p) {
    return (unsigned)(size_t)p;
}

// ---------------- conversion kernels ----------------

__global__ void cvt_f32_to_bf16(const float* __restrict__ src,
                                __bf16* __restrict__ dst, int n) {
    int i = blockIdx.x * blockDim.x + threadIdx.x;
    if (i < n) dst[i] = (__bf16)src[i];
}

// src: K x N row-major f32  ->  dst: N x Kp row-major bf16, zero pad k>=K
__global__ void cvt_transpose_pad(const float* __restrict__ src,
                                  __bf16* __restrict__ dst,
                                  int K, int N, int Kp) {
    int i = blockIdx.x * blockDim.x + threadIdx.x;
    if (i >= N * Kp) return;
    int n = i / Kp, k = i % Kp;
    float v = (k < K) ? src[(size_t)k * N + n] : 0.0f;
    dst[i] = (__bf16)v;
}

// x_dbl (M x 80) cols [0,48) -> (M x 64) bf16, zero pad
__global__ void extract_delta_bf16(const float* __restrict__ x_dbl,
                                   __bf16* __restrict__ dst, int M) {
    int i = blockIdx.x * blockDim.x + threadIdx.x;
    if (i >= M * KPAD_DT) return;
    int m = i / KPAD_DT, k = i % KPAD_DT;
    float v = (k < DT_RANK) ? x_dbl[(size_t)m * XDBL_W + k] : 0.0f;
    dst[i] = (__bf16)v;
}

// ---------------- epilogue helper ----------------

__device__ __forceinline__ float apply_epilogue(float v, const float* bias,
                                                int col, int epilogue) {
    if (epilogue == 1) {
        v += bias[col];
        v = (v > 20.0f) ? v : log1pf(__expf(v));   // softplus
    }
    return v;
}

// ---------------- WMMA GEMM, direct-global B (small N) ----------------
// C[M x N] f32 = A[M x K] bf16 * B ; Bt[N x K] bf16 is B^T.
// One wave -> 16 x (16*NT) tile. K%32==0, M%16==0, N%(16*NT)==0.
template <int NT>
__global__ void gemm_bf16_wmma(const __bf16* __restrict__ A,
                               const __bf16* __restrict__ Bt,
                               float* __restrict__ C,
                               int M, int N, int K,
                               const float* __restrict__ bias, int epilogue) {
    const int lane   = threadIdx.x & 31;
    const int lanem  = lane & 15;
    const int lhalf  = lane >> 4;
    const int wave   = (blockIdx.x * blockDim.x + threadIdx.x) >> 5;

    const int tilesN = N / (16 * NT);
    const int tm = wave / tilesN;
    const int tn = wave % tilesN;
    if (tm * 16 >= M) return;

    const int row0 = tm * 16;
    const int col0 = tn * 16 * NT;

    const int kh8 = lhalf * 8;            // A-frag K: k = 16*lhalf + s (contiguous order)
    const unsigned int* arow =
        reinterpret_cast<const unsigned int*>(A + (size_t)(row0 + lanem) * K);

    const unsigned int* brow[NT];
#pragma unroll
    for (int t = 0; t < NT; ++t)
        brow[t] = reinterpret_cast<const unsigned int*>(
            Bt + (size_t)(col0 + t * 16 + lanem) * K);
    const int khB_dw = lhalf * 8;

    v8f acc[NT] = {};

    for (int kb = 0; kb < K; kb += 32) {
        Frag af;
#pragma unroll
        for (int v = 0; v < 8; ++v) {
            int kk = 2 * kh8 + 2 * v;
            af.u[v] = arow[(kb + kk) >> 1];
        }
#pragma unroll
        for (int t = 0; t < NT; ++t) {
            Frag bf;
#pragma unroll
            for (int v = 0; v < 8; ++v)
                bf.u[v] = brow[t][(kb >> 1) + khB_dw + v];
            acc[t] = __builtin_amdgcn_wmma_f32_16x16x32_bf16(
                false, af.v, false, bf.v, (short)0, acc[t], false, false);
        }
    }

    const int rbase = row0 + (lhalf << 3);
#pragma unroll
    for (int t = 0; t < NT; ++t) {
        const int col = col0 + t * 16 + lanem;
#pragma unroll
        for (int r = 0; r < 8; ++r)
            C[(size_t)(rbase + r) * N + col] =
                apply_epilogue(acc[t][r], bias, col, epilogue);
    }
}

// ---------------- WMMA GEMM, async-LDS-staged B ----------------
// Block = 128 threads (4 waves) -> 64x64 block tile; each wave owns a 16-row
// stripe, all four share the 64-column B panel staged in LDS by
// global_load_async_to_lds_b128 (double-buffered, 1 barrier / k-step).
// LDS panel layout: per column c in [0,64): 64 contiguous bytes = K 0..31 bf16.

__device__ __forceinline__ void issue_b_panel(const __bf16* __restrict__ Bt,
                                              int K, int col0, int kb,
                                              unsigned char* buf, int tid) {
    const char* gbase = reinterpret_cast<const char*>(Bt);
#pragma unroll
    for (int ch = tid; ch < 256; ch += 128) {          // 256 x 16B = 4KB
        int c   = ch >> 2;                             // column 0..63
        int sub = ch & 3;                              // 16B chunk in column
        const void* g = gbase + ((size_t)(col0 + c) * K + kb) * 2 + sub * 16;
        async_g2l_b128(lds_off_of(buf + ch * 16), g);
    }
}

__global__ void __launch_bounds__(128)
gemm_bf16_wmma_lds(const __bf16* __restrict__ A,
                   const __bf16* __restrict__ Bt,
                   float* __restrict__ C,
                   int M, int N, int K,
                   const float* __restrict__ bias, int epilogue) {
    __shared__ __align__(16) unsigned char smem[2][64 * 64];   // 2 x 4KB

    const int tid   = threadIdx.x;
    const int lane  = tid & 31;
    const int lanem = lane & 15;
    const int lhalf = lane >> 4;
    const int wid   = tid >> 5;

    const int nb   = N / 64;
    const int bm   = blockIdx.x / nb;
    const int bn   = blockIdx.x % nb;
    const int row0 = bm * 64 + wid * 16;
    const int col0 = bn * 64;

    const int kh8 = lhalf * 8;
    const unsigned int* arow =
        reinterpret_cast<const unsigned int*>(A + (size_t)(row0 + lanem) * K);

    v8f acc[4] = {};

    issue_b_panel(Bt, K, col0, 0, smem[0], tid);

    const int ksteps = K / 32;
    for (int s = 0; s < ksteps; ++s) {
        const int kb = s * 32;
        wait_asynccnt0();        // our panel transfers landed in LDS
        __syncthreads();         // all waves' transfers visible; prev reads done

        if (s + 1 < ksteps)
            issue_b_panel(Bt, K, col0, kb + 32, smem[(s + 1) & 1], tid);

        Frag af;
#pragma unroll
        for (int v = 0; v < 8; ++v) {
            int kk = 2 * kh8 + 2 * v;
            af.u[v] = arow[(kb + kk) >> 1];
        }

        const unsigned char* bufp = smem[s & 1];
#pragma unroll
        for (int t = 0; t < 4; ++t) {
            const unsigned int* bsm = reinterpret_cast<const unsigned int*>(
                bufp + (size_t)(t * 16 + lanem) * 64 + lhalf * 32);
            Frag bf;
#pragma unroll
            for (int v = 0; v < 8; ++v) bf.u[v] = bsm[v];
            acc[t] = __builtin_amdgcn_wmma_f32_16x16x32_bf16(
                false, af.v, false, bf.v, (short)0, acc[t], false, false);
        }
    }

    const int rbase = row0 + (lhalf << 3);
#pragma unroll
    for (int t = 0; t < 4; ++t) {
        const int col = col0 + t * 16 + lanem;
#pragma unroll
        for (int r = 0; r < 8; ++r)
            C[(size_t)(rbase + r) * N + col] =
                apply_epilogue(acc[t][r], bias, col, epilogue);
    }
}

// ---------------- depthwise causal conv + SiLU ----------------

__global__ void conv_silu_kernel(const float* __restrict__ x_and_res,
                                 const float* __restrict__ conv_w,
                                 const float* __restrict__ conv_b,
                                 float* __restrict__ xc_f32,
                                 __bf16* __restrict__ xc_bf) {
    int i = blockIdx.x * blockDim.x + threadIdx.x;
    if (i >= B_SZ * LSEQ * D_IN) return;
    int d  = i % D_IN;
    int bl = i / D_IN;
    int l  = bl % LSEQ;
    int b  = bl / LSEQ;

    float acc = conv_b[d];
#pragma unroll
    for (int j = 0; j < KCONV; ++j) {
        int ls = l - (KCONV - 1) + j;
        if (ls >= 0)
            acc += x_and_res[(size_t)(b * LSEQ + ls) * (2 * D_IN) + d] *
                   conv_w[j * D_IN + d];
    }
    float s = acc / (1.0f + __expf(-acc));
    xc_f32[i] = s;
    xc_bf[i]  = (__bf16)s;
}

// ---------------- fused selective scan ----------------
// h_t = exp(delta_t*A) h_{t-1} + delta_t*u_t*B_t ; y = sum_n h*C + u*D,
// gated by silu(res). Lane = (d_sub, n); 16-lane shfl_xor reduction over n.

__global__ void scan_kernel(const float* __restrict__ xc,
                            const float* __restrict__ delta,
                            const float* __restrict__ x_dbl,
                            const float* __restrict__ A_log,
                            const float* __restrict__ D_param,
                            const float* __restrict__ x_and_res,
                            __bf16* __restrict__ ys_bf) {
    const int n    = threadIdx.x & 15;
    const int dsub = threadIdx.x >> 4;
    const int b    = blockIdx.x / (D_IN / 16);
    const int d    = (blockIdx.x % (D_IN / 16)) * 16 + dsub;

    const float Ad = -__expf(A_log[d * NSTATE + n]);
    const float Dp = D_param[d];
    float h = 0.0f;

    for (int l = 0; l < LSEQ; ++l) {
        const int bl = b * LSEQ + l;
        const float dv = delta[(size_t)bl * D_IN + d];
        const float uv = xc[(size_t)bl * D_IN + d];
        const float Bv = x_dbl[(size_t)bl * XDBL_W + DT_RANK + n];
        const float Cv = x_dbl[(size_t)bl * XDBL_W + DT_RANK + NSTATE + n];

        h = __expf(dv * Ad) * h + dv * uv * Bv;

        float contrib = h * Cv;
#pragma unroll
        for (int off = 8; off > 0; off >>= 1)
            contrib += __shfl_xor(contrib, off, 16);

        if (n == 0) {
            float y = contrib + uv * Dp;
            float r = x_and_res[(size_t)bl * (2 * D_IN) + D_IN + d];
            float g = r / (1.0f + __expf(-r));
            ys_bf[(size_t)bl * D_IN + d] = (__bf16)(y * g);
        }
    }
}

// ---------------- host side ----------------

static inline size_t align256(size_t x) { return (x + 255) & ~(size_t)255; }

extern "C" void kernel_launch(void* const* d_in, const int* in_sizes, int n_in,
                              void* d_out, int out_size, void* d_ws, size_t ws_size,
                              hipStream_t stream) {
    const float* x      = (const float*)d_in[0];
    const float* W_in   = (const float*)d_in[1];
    const float* conv_w = (const float*)d_in[2];
    const float* conv_b = (const float*)d_in[3];
    const float* W_x    = (const float*)d_in[4];
    const float* W_dt   = (const float*)d_in[5];
    const float* b_dt   = (const float*)d_in[6];
    const float* A_log  = (const float*)d_in[7];
    const float* D_par  = (const float*)d_in[8];
    const float* W_out  = (const float*)d_in[9];
    float* out = (float*)d_out;

    char* p = (char*)d_ws;
    size_t off = 0;
    auto carve = [&](size_t bytes) { void* r = p + off; off += align256(bytes); return r; };

    __bf16* W_in_t  = (__bf16*)carve((size_t)3072 * 768 * 2);
    __bf16* W_x_t   = (__bf16*)carve((size_t)XDBL_W * 1536 * 2);
    __bf16* W_dt_t  = (__bf16*)carve((size_t)1536 * KPAD_DT * 2);
    __bf16* W_out_t = (__bf16*)carve((size_t)768 * 1536 * 2);
    __bf16* x_bf    = (__bf16*)carve((size_t)MROWS * D_MODEL * 2);
    float*  xr      = (float*) carve((size_t)MROWS * 2 * D_IN * 4);
    float*  xc_f32  = (float*) carve((size_t)MROWS * D_IN * 4);
    __bf16* xc_bf   = (__bf16*)carve((size_t)MROWS * D_IN * 2);
    float*  x_dbl   = (float*) carve((size_t)MROWS * XDBL_W * 4);
    __bf16* dlt_bf  = (__bf16*)carve((size_t)MROWS * KPAD_DT * 2);
    float*  dlt_f32 = (float*) carve((size_t)MROWS * D_IN * 4);
    __bf16* ys_bf   = (__bf16*)carve((size_t)MROWS * D_IN * 2);
    (void)ws_size;

    const int TB = 256;
    auto blocks = [](int n, int tb) { return (n + tb - 1) / tb; };

    // 0) conversions / transposes
    cvt_f32_to_bf16<<<blocks(MROWS * D_MODEL, TB), TB, 0, stream>>>(x, x_bf, MROWS * D_MODEL);
    cvt_transpose_pad<<<blocks(3072 * 768, TB), TB, 0, stream>>>(W_in, W_in_t, 768, 3072, 768);
    cvt_transpose_pad<<<blocks(XDBL_W * 1536, TB), TB, 0, stream>>>(W_x, W_x_t, 1536, XDBL_W, 1536);
    cvt_transpose_pad<<<blocks(1536 * KPAD_DT, TB), TB, 0, stream>>>(W_dt, W_dt_t, DT_RANK, 1536, KPAD_DT);
    cvt_transpose_pad<<<blocks(768 * 1536, TB), TB, 0, stream>>>(W_out, W_out_t, 1536, 768, 1536);

    // 1) x @ W_in -> x_and_res  (M=2048, N=3072, K=768), LDS-staged B
    gemm_bf16_wmma_lds<<<(MROWS / 64) * (3072 / 64), 128, 0, stream>>>(
        x_bf, W_in_t, xr, MROWS, 3072, 768, nullptr, 0);

    // 2) depthwise causal conv + SiLU
    conv_silu_kernel<<<blocks(B_SZ * LSEQ * D_IN, TB), TB, 0, stream>>>(
        xr, conv_w, conv_b, xc_f32, xc_bf);

    // 3) xc @ W_x -> x_dbl  (M=2048, N=80, K=1536), direct-global (NT=5)
    {
        int waves = (MROWS / 16) * (XDBL_W / 80);
        gemm_bf16_wmma<5><<<waves / 4, 128, 0, stream>>>(
            xc_bf, W_x_t, x_dbl, MROWS, XDBL_W, 1536, nullptr, 0);
    }

    // 4) delta slice -> bf16, K padded 48 -> 64
    extract_delta_bf16<<<blocks(MROWS * KPAD_DT, TB), TB, 0, stream>>>(x_dbl, dlt_bf, MROWS);

    // 5) delta @ W_dt + b_dt -> softplus  (M=2048, N=1536, K=64), LDS-staged
    gemm_bf16_wmma_lds<<<(MROWS / 64) * (1536 / 64), 128, 0, stream>>>(
        dlt_bf, W_dt_t, dlt_f32, MROWS, 1536, 64, b_dt, 1);

    // 6) fused selective scan + D skip + silu(res) gate -> ys (bf16)
    scan_kernel<<<B_SZ * (D_IN / 16), 256, 0, stream>>>(
        xc_f32, dlt_f32, x_dbl, A_log, D_par, xr, ys_bf);

    // 7) ys @ W_out -> out  (M=2048, N=768, K=1536), LDS-staged B
    gemm_bf16_wmma_lds<<<(MROWS / 64) * (768 / 64), 128, 0, stream>>>(
        ys_bf, W_out_t, out, MROWS, 768, 1536, nullptr, 0);
}
